// DigitCaps_75574244540483
// MI455X (gfx1250) — matrix operand order repaired and checked by reading, not compiled, not run on hardware
//
#include <hip/hip_runtime.h>
#include <hip/hip_bf16.h>

typedef _Float16 v16h __attribute__((ext_vector_type(16)));
typedef _Float16 v8h  __attribute__((ext_vector_type(8)));
typedef float    v8f  __attribute__((ext_vector_type(8)));

#define B_   64
#define I_   4096
#define P_   8          // Din
#define N_   10
#define D_   16
#define KTOT (I_ * P_)  // 32768
#define CTOT (N_ * D_)  // 160
#define EPS_ 1e-7f

// ---------- K0a: inputs fp32 [b][i][p] -> fp16 A [b][k], k = i*8+p ----------
__global__ void conv_a_kernel(const float* __restrict__ in, _Float16* __restrict__ out, int n) {
  int idx = blockIdx.x * blockDim.x + threadIdx.x;
  if (idx < n) out[idx] = (_Float16)in[idx];
}

// ---------- K0b: W fp32 [i][n][p][d] -> fp16 Bt [c][k], c=n*16+d, k=i*8+p ----------
__global__ void conv_b_kernel(const float* __restrict__ W, _Float16* __restrict__ Bt) {
  int idx = blockIdx.x * blockDim.x + threadIdx.x;
  if (idx >= I_ * N_ * P_ * D_) return;
  int i  = idx / (N_ * P_ * D_);          // /1280
  int r  = idx - i * (N_ * P_ * D_);
  int n  = r >> 7;                        // /128
  int r2 = r & 127;
  int p  = r2 >> 4;
  int d  = r2 & 15;
  int c  = n * 16 + d;
  int k  = i * 8 + p;
  Bt[(size_t)c * KTOT + k] = (_Float16)W[idx];
}

// ---------- K1: WMMA GEMM  s0[b][c] += sum_k A[b][k] * Bt[c][k] ----------
// grid.x = KTOT/512 k-chunks; 8 waves per block cover 4 m-tiles x 10 n-tiles.
__global__ void gemm_wmma_kernel(const _Float16* __restrict__ A,
                                 const _Float16* __restrict__ Bt,
                                 float* __restrict__ s_accum) {
  const int lane   = threadIdx.x & 31;
  const int wave   = threadIdx.x >> 5;
  const int m_tile = wave & 3;             // 4 tiles of 16 rows (b)
  const int nbase  = (wave >> 2) * 5;      // 5 n-tiles per wave
  const int k0     = blockIdx.x * 512;
  const int hi     = lane >> 4;            // half-wave selector
  const int lcol   = lane & 15;

  v8f acc[5] = {};

  const int row   = m_tile * 16 + lcol;    // A row (b)
  const int koffA = hi * 16;               // lanes 16-31 start at K+16

  for (int ks = 0; ks < 512; ks += 32) {
    const int kb = k0 + ks;
    // A fragment: elems 0..15 = K[koffA..koffA+15]
    const _Float16* pa = A + (size_t)row * KTOT + kb + koffA;
    v8h alo = *(const v8h*)(pa);
    v8h ahi = *(const v8h*)(pa + 8);
    v16h afrag = __builtin_shufflevector(alo, ahi,
        0, 1, 2, 3, 4, 5, 6, 7, 8, 9, 10, 11, 12, 13, 14, 15);
#pragma unroll
    for (int t = 0; t < 5; ++t) {
      // B fragment: lane holds column (lane&15); lanes 0-15 K=kb..kb+15, 16-31 K=kb+16..kb+31
      const int col = (nbase + t) * 16 + lcol;
      const _Float16* pb = Bt + (size_t)col * KTOT + kb + hi * 16;
      v16h bfrag = *(const v16h*)pb;
      acc[t] = __builtin_amdgcn_wmma_f32_16x16x32_f16(
          /*neg_a=*/false, afrag, /*neg_b=*/false, bfrag,
          /*c_mod=*/(short)0, acc[t], /*reuse_a=*/false, /*reuse_b=*/false);
    }
  }

#pragma unroll
  for (int t = 0; t < 5; ++t) {
    const int colg = (nbase + t) * 16 + lcol;
#pragma unroll
    for (int r = 0; r < 8; ++r) {
      int bg = m_tile * 16 + r + hi * 8;   // C layout: VGPR r -> M=r (lanes<16) / M=r+8
      atomicAdd(&s_accum[bg * CTOT + colg], acc[t][r]);
    }
  }
}

// ---------- K2: elementwise squash (the axis=-2 reduce is over a size-1 axis) ----------
__global__ void squash_kernel(const float* __restrict__ s, float scale,
                              float* __restrict__ v_out, float* __restrict__ v_acc) {
  int idx = blockIdx.x * blockDim.x + threadIdx.x;
  if (idx >= B_ * CTOT) return;
  float sv = s[idx] * scale;
  float sq = sv * sv;
  float f  = sq / (1.0f + sq) / sqrtf(sq + EPS_);
  float v  = f * sv;
  v_out[idx] = v;
  if (v_acc) v_acc[idx] += v;
}

// ---------- K3: fused routing pass (recompute u_hat; logits use running Vacc) ----------
// s_out[b][n][d] += softmax_n( u_hat[b,i,n,:] . V[b,n,:] ) * u_hat[b,i,n,d], summed over i.
__global__ void routing_kernel(const float* __restrict__ inputs,
                               const float* __restrict__ W,
                               const float* __restrict__ V,
                               float* __restrict__ s_out) {
  __shared__ float Vs[CTOT];
  __shared__ float s_block[CTOT];
  const int b = blockIdx.y;
  const int i = blockIdx.x * blockDim.x + threadIdx.x;
  if (threadIdx.x < CTOT) {
    Vs[threadIdx.x] = V[b * CTOT + threadIdx.x];
    s_block[threadIdx.x] = 0.0f;
  }
  __syncthreads();

  float xr[8];
  const float* x = inputs + ((size_t)b * I_ + i) * P_;
#pragma unroll
  for (int p = 0; p < 8; ++p) xr[p] = x[p];

  const float* Wi = W + (size_t)i * (N_ * P_ * D_);
  __builtin_prefetch(Wi, 0, 0);   // global_prefetch_b8

  // pass 1: logits t[n] = sum_d u_hat[n][d] * Vs[n][d]
  float t[10];
#pragma unroll
  for (int n = 0; n < 10; ++n) {
    const float* Wn = Wi + n * (P_ * D_);
    float accn = 0.0f;
#pragma unroll
    for (int d = 0; d < 16; ++d) {
      float uh = 0.0f;
#pragma unroll
      for (int p = 0; p < 8; ++p) uh = fmaf(xr[p], Wn[p * 16 + d], uh);
      accn = fmaf(uh, Vs[n * 16 + d], accn);
    }
    t[n] = accn;
  }

  // softmax over n (axis=2)
  float m = t[0];
#pragma unroll
  for (int n = 1; n < 10; ++n) m = fmaxf(m, t[n]);
  float c[10];
  float sum = 0.0f;
#pragma unroll
  for (int n = 0; n < 10; ++n) { c[n] = __expf(t[n] - m); sum += c[n]; }
  const float inv = 1.0f / sum;

  // pass 2: recompute u_hat, accumulate c[n]*u_hat into LDS (ds_add_f32)
#pragma unroll
  for (int n = 0; n < 10; ++n) {
    const float* Wn = Wi + n * (P_ * D_);
    const float cn = c[n] * inv;
#pragma unroll
    for (int d = 0; d < 16; ++d) {
      float uh = 0.0f;
#pragma unroll
      for (int p = 0; p < 8; ++p) uh = fmaf(xr[p], Wn[p * 16 + d], uh);
      atomicAdd(&s_block[n * 16 + d], cn * uh);
    }
  }
  __syncthreads();
  if (threadIdx.x < CTOT)
    atomicAdd(&s_out[b * CTOT + threadIdx.x], s_block[threadIdx.x]);
}

extern "C" void kernel_launch(void* const* d_in, const int* in_sizes, int n_in,
                              void* d_out, int out_size, void* d_ws, size_t ws_size,
                              hipStream_t stream) {
  const float* inputs = (const float*)d_in[0];  // [64, 4096, 8] fp32
  const float* W      = (const float*)d_in[1];  // [4096, 10, 8, 16] fp32

  char* ws = (char*)d_ws;
  _Float16* A_h  = (_Float16*)ws;                                   // 4 MB
  _Float16* Bt_h = (_Float16*)(ws + (size_t)B_ * KTOT * 2);         // 10.5 MB
  float* F = (float*)(ws + (size_t)B_ * KTOT * 2 + (size_t)CTOT * KTOT * 2);
  float* s0   = F;
  float* s1   = F + 1 * B_ * CTOT;
  float* s2   = F + 2 * B_ * CTOT;
  float* Vacc = F + 3 * B_ * CTOT;   // running v0 (+ v1)
  float* vtmp = F + 4 * B_ * CTOT;

  // zero all atomic-accumulated buffers every call (deterministic)
  hipMemsetAsync(F, 0, (size_t)4 * B_ * CTOT * sizeof(float), stream);

  conv_a_kernel<<<(B_ * KTOT + 255) / 256, 256, 0, stream>>>(inputs, A_h, B_ * KTOT);
  conv_b_kernel<<<(I_ * N_ * P_ * D_ + 255) / 256, 256, 0, stream>>>(W, Bt_h);

  // iter 0: c = 1/N uniform -> s0 = (1/N) * GEMM over K = I*Din
  gemm_wmma_kernel<<<KTOT / 512, 256, 0, stream>>>(A_h, Bt_h, s0);
  squash_kernel<<<(B_ * CTOT + 255) / 256, 256, 0, stream>>>(s0, 1.0f / N_, vtmp, Vacc); // Vacc = v0

  // iter 1: logits = u_hat . v0
  routing_kernel<<<dim3(I_ / 256, B_), 256, 0, stream>>>(inputs, W, Vacc, s1);
  squash_kernel<<<(B_ * CTOT + 255) / 256, 256, 0, stream>>>(s1, 1.0f, vtmp, Vacc);      // Vacc = v0+v1

  // iter 2: logits = u_hat . (v0+v1)  (b accumulates across iterations)
  routing_kernel<<<dim3(I_ / 256, B_), 256, 0, stream>>>(inputs, W, Vacc, s2);
  squash_kernel<<<(B_ * CTOT + 255) / 256, 256, 0, stream>>>(s2, 1.0f, (float*)d_out, nullptr);
}
